// BILSTM_CRF_27462020890810
// MI455X (gfx1250) — matrix operand order repaired and checked by reading, not compiled, not run on hardware
//
#include <hip/hip_runtime.h>
#include <hip/hip_bf16.h>

// ---------------------------------------------------------------------------
// BiLSTM-CRF for MI455X (gfx1250, wave32, WMMA + async global->LDS staging)
// ---------------------------------------------------------------------------

typedef _Float16 h16;
typedef __attribute__((ext_vector_type(8)))  _Float16 v8h;
typedef __attribute__((ext_vector_type(16))) _Float16 v16h;
typedef __attribute__((ext_vector_type(8)))  float    v8f;

#define BB 256
#define TT 1024
#define DIN0 39
#define HH 64      // per-direction hidden
#define GG 256     // 4*HH gates
#define CC 41      // tags
#define CP 48      // padded tag dim

__device__ __forceinline__ float fsig(float x) {
    return 1.0f / (1.0f + __expf(-x));
}
__device__ __forceinline__ float ftanh(float x) {
    float e = __expf(-2.0f * fabsf(x));
    float r = (1.0f - e) / (1.0f + e);
    return copysignf(r, x);
}

// Build A-fragment (16x32 f16, contiguous k order) from a row pointer.
// lane row m = lane&15, hi = lane>>4:
//   halves 0..15 = K[kbase + hi*16 .. +15]
__device__ __forceinline__ v16h load_a_frag(const h16* row, int kbase, int hi) {
    v8h lo = *(const v8h*)(row + kbase + hi * 16);
    v8h hh = *(const v8h*)(row + kbase + hi * 16 + 8);
    return __builtin_shufflevector(lo, hh, 0,1,2,3,4,5,6,7,8,9,10,11,12,13,14,15);
}

// ---------------------------------------------------------------------------
// Conversion / padding kernels (f32 -> f16)
// ---------------------------------------------------------------------------
__global__ void pad_x_kernel(const float* __restrict__ x, h16* __restrict__ xp, int total) {
    int i = blockIdx.x * blockDim.x + threadIdx.x;
    if (i >= total) return;
    int bt = i >> 6, k = i & 63;                  // [B*T, 64] padded
    xp[i] = (h16)(k < DIN0 ? x[bt * DIN0 + k] : 0.0f);
}

__global__ void cvt_kernel(const float* __restrict__ s, h16* __restrict__ d, int n) {
    int i = blockIdx.x * blockDim.x + threadIdx.x;
    if (i < n) d[i] = (h16)s[i];
}

__global__ void pad_wih0_kernel(const float* __restrict__ s, h16* __restrict__ d, int total) {
    int i = blockIdx.x * blockDim.x + threadIdx.x;
    if (i >= total) return;                       // total = 2*256*64
    int k = i & 63, dg = i >> 6;                  // dg over [2*256]
    d[i] = (h16)(k < DIN0 ? s[dg * DIN0 + k] : 0.0f);
}

__global__ void pad_linw_kernel(const float* __restrict__ s, h16* __restrict__ d) {
    int i = blockIdx.x * blockDim.x + threadIdx.x;
    if (i >= CP * 128) return;
    int r = i >> 7, k = i & 127;                  // [48,128], rows >=41 zero
    d[i] = (h16)(r < CC ? s[r * 128 + k] : 0.0f);
}

// ---------------------------------------------------------------------------
// LSTM scan: one (layer, direction) per blockIdx.y, batch tile of 16 per
// blockIdx.x.  512 threads = 16 waves; wave w owns gate columns [16w,16w+16).
// x tiles are double-buffered in LDS via async global->LDS loads issued by
// wave 0 one timestep ahead (ASYNCcnt pipelining).
// ---------------------------------------------------------------------------
template <int DIN>
__global__ void __launch_bounds__(512)
lstm_scan_kernel(const h16* __restrict__ xin,   // [B,T,DIN] f16
                 const h16* __restrict__ wih,   // [2,G,DIN] f16
                 const h16* __restrict__ whh,   // [2,G,64]  f16
                 const float* __restrict__ bias,// [2,G]     f32 (fused b_ih+b_hh)
                 h16* __restrict__ hout)        // [B,T,128] f16 (dir -> cols)
{
    constexpr int KF      = DIN / 32;
    constexpr int ROWB    = DIN * 2;        // bytes per tile row
    constexpr int NI      = (16 * ROWB) / (32 * 16); // async b128 issues per tile (4 or 8)
    const int dir  = blockIdx.y;
    const int b0   = blockIdx.x * 16;
    const int tid  = threadIdx.x;
    const int wave = tid >> 5;        // N-tile 0..15
    const int lane = tid & 31;
    const int lm   = lane & 15;
    const int hi   = lane >> 4;

    __shared__ __align__(16) float gatesLDS[16][GG];   // 16 KB
    __shared__ __align__(16) float cLDS[16][HH];       // 4 KB
    __shared__ __align__(16) h16   hLDS[16][HH];       // 2 KB
    __shared__ __align__(16) h16   xLDS[2][16][DIN];   // 4/8 KB double buffer

    // Preload loop-invariant weight B-fragments (B^T column n = weight row n).
    const int ncol = wave * 16 + lm;
    v16h bwih[KF], bwhh[2];
    {
        const h16* wr = wih + ((size_t)dir * GG + ncol) * DIN;
        #pragma unroll
        for (int k = 0; k < KF; ++k) bwih[k] = *(const v16h*)(wr + k * 32 + hi * 16);
        const h16* wr2 = whh + ((size_t)dir * GG + ncol) * HH;
        #pragma unroll
        for (int k = 0; k < 2; ++k) bwhh[k] = *(const v16h*)(wr2 + k * 32 + hi * 16);
    }
    const float bval = bias[dir * GG + ncol];

    // Per-lane async-copy addressing (wave 0 only): 16B chunk per lane/issue.
    size_t   gbase[NI];   // global half-index base, add t*DIN per step
    uint32_t lbase[NI];   // LDS byte offset within one x buffer
    #pragma unroll
    for (int it = 0; it < NI; ++it) {
        int li  = (it * 32 + lane) * 16;  // byte index within 16xDIN tile
        int row = li / ROWB;
        int cb  = li % ROWB;
        gbase[it] = ((size_t)(b0 + row)) * TT * DIN + cb / 2;
        lbase[it] = (uint32_t)(size_t)(&xLDS[0][row][0]) + (uint32_t)cb;
    }
    const uint32_t xbufBytes = (uint32_t)sizeof(h16) * 16 * DIN;

    // zero-init h and c
    for (int e = tid; e < 16 * HH; e += 512) {
        int m = e >> 6, n = e & 63;
        cLDS[m][n] = 0.0f;
        hLDS[m][n] = (h16)0.0f;
    }

    // Prologue: stage x tile for step 0 into buffer 0.
    if (wave == 0) {
        const int t0 = dir ? (TT - 1) : 0;
        #pragma unroll
        for (int it = 0; it < NI; ++it) {
            uint64_t ga = (uint64_t)(size_t)(xin + gbase[it] + (size_t)t0 * DIN);
            uint32_t la = lbase[it];
            asm volatile("global_load_async_to_lds_b128 %0, %1, off"
                         :: "v"(la), "v"(ga) : "memory");
        }
    }

    for (int s = 0; s < TT; ++s) {
        const int t  = dir ? (TT - 1 - s) : s;
        const int bb = s & 1;

        if (wave == 0) {
            if (s + 1 < TT) {
                // issue next tile into the other buffer, then wait until only
                // that batch is outstanding (=> current buffer complete)
                const int tn = dir ? (t - 1) : (t + 1);
                #pragma unroll
                for (int it = 0; it < NI; ++it) {
                    uint64_t ga = (uint64_t)(size_t)(xin + gbase[it] + (size_t)tn * DIN);
                    uint32_t la = lbase[it] + (uint32_t)(bb ^ 1) * xbufBytes;
                    asm volatile("global_load_async_to_lds_b128 %0, %1, off"
                                 :: "v"(la), "v"(ga) : "memory");
                }
                if constexpr (DIN == 64) {
                    asm volatile("s_wait_asynccnt 0x4" ::: "memory");
                } else {
                    asm volatile("s_wait_asynccnt 0x8" ::: "memory");
                }
                // warm GL2 two steps ahead
                if (s + 2 < TT) {
                    const int t2 = dir ? (t - 2) : (t + 2);
                    __builtin_prefetch(xin + gbase[0] + (size_t)t2 * DIN, 0, 0);
                }
            } else {
                asm volatile("s_wait_asynccnt 0x0" ::: "memory");
            }
        }
        __syncthreads();   // x[bb] staged, h from previous step visible

        v8f acc = { bval, bval, bval, bval, bval, bval, bval, bval };

        // --- input projection: [16,DIN] x [DIN,16] from LDS ---
        const h16* arow = &xLDS[bb][lm][0];
        #pragma unroll
        for (int k = 0; k < KF; ++k) {
            v16h a = load_a_frag(arow, k * 32, hi);
            acc = __builtin_amdgcn_wmma_f32_16x16x32_f16(
                false, a, false, bwih[k], (short)0, acc, false, false);
        }

        // --- recurrence: [16,64] x [64,16] from LDS h ---
        const h16* hrow = &hLDS[lm][0];
        #pragma unroll
        for (int k = 0; k < 2; ++k) {
            v16h a = load_a_frag(hrow, k * 32, hi);
            acc = __builtin_amdgcn_wmma_f32_16x16x32_f16(
                false, a, false, bwhh[k], (short)0, acc, false, false);
        }

        // scatter gate tile to LDS (D layout: VGPR r -> M = r + hi*8, N = lm)
        #pragma unroll
        for (int r = 0; r < 8; ++r)
            gatesLDS[r + hi * 8][wave * 16 + lm] = acc[r];
        __syncthreads();

        // elementwise LSTM cell update (torch gate order i,f,g,o).
        // Each thread owns one (m, n-pair): 512 threads cover 16x32 pairs.
        {
            const int m  = tid >> 5;
            const int np = (tid & 31) * 2;
            float h2[2];
            #pragma unroll
            for (int j = 0; j < 2; ++j) {
                const int n = np + j;
                float gi = fsig(gatesLDS[m][n]);
                float gf = fsig(gatesLDS[m][HH + n]);
                float gg = ftanh(gatesLDS[m][2 * HH + n]);
                float go = fsig(gatesLDS[m][3 * HH + n]);
                float c  = gf * cLDS[m][n] + gi * gg;
                cLDS[m][n] = c;
                h2[j] = go * ftanh(c);
            }
            h16 hp0 = (h16)h2[0], hp1 = (h16)h2[1];
            uint32_t packed = (uint32_t)__builtin_bit_cast(unsigned short, hp0) |
                              ((uint32_t)__builtin_bit_cast(unsigned short, hp1) << 16);
            *(uint32_t*)(&hLDS[m][np]) = packed;
            *(uint32_t*)(&hout[(((size_t)(b0 + m)) * TT + t) * 128 + dir * HH + np]) = packed;
        }
        // next iteration's top barrier orders hLDS writes vs reads
    }
}

// ---------------------------------------------------------------------------
// Emissions: [B*T,128] x [128,48] (rows >=41 of W zero-padded) + bias.
// 256 threads = 8 wave-tasks per block; task = (mtile, ntile).
// ---------------------------------------------------------------------------
__global__ void __launch_bounds__(256)
emis_kernel(const h16* __restrict__ feats,  // [B*T,128] f16
            const h16* __restrict__ linw,   // [48,128]  f16
            const float* __restrict__ linb, // [41]      f32
            float* __restrict__ ems)        // [B*T,48]  f32
{
    const int tid  = threadIdx.x;
    const int wave = tid >> 5;
    const int lane = tid & 31;
    const int lm   = lane & 15;
    const int hi   = lane >> 4;

    const int wg    = blockIdx.x * 8 + wave;   // 0 .. 49151
    const int mtile = wg / 3;                  // 0 .. 16383
    const int ntile = wg % 3;
    const int n     = ntile * 16 + lm;

    v16h bw[4];
    const h16* wr = linw + (size_t)n * 128;
    #pragma unroll
    for (int k = 0; k < 4; ++k) bw[k] = *(const v16h*)(wr + k * 32 + hi * 16);

    const float bval = (n < CC) ? linb[n] : -1.0e30f;
    v8f acc = { bval, bval, bval, bval, bval, bval, bval, bval };

    const h16* arow = feats + ((size_t)(mtile * 16 + lm)) * 128;
    #pragma unroll
    for (int k = 0; k < 4; ++k) {
        v16h a = load_a_frag(arow, k * 32, hi);
        acc = __builtin_amdgcn_wmma_f32_16x16x32_f16(
            false, a, false, bw[k], (short)0, acc, false, false);
    }

    #pragma unroll
    for (int r = 0; r < 8; ++r)
        ems[((size_t)(mtile * 16) + r + hi * 8) * CP + n] = acc[r];
}

// ---------------------------------------------------------------------------
// Viterbi: one block (64 threads) per batch row; trans cached in LDS,
// double-buffered scores, u8 backpointers, serial backtrace by thread 0.
// ---------------------------------------------------------------------------
__global__ void __launch_bounds__(64)
viterbi_kernel(const float* __restrict__ ems,    // [B*T,48]
               const float* __restrict__ startv, // [41]
               const float* __restrict__ endv,   // [41]
               const float* __restrict__ trans,  // [41,41]
               unsigned char* __restrict__ bp,   // [B, T-1, 48]
               int* __restrict__ out)            // [B,T]
{
    const int b   = blockIdx.x;
    const int tid = threadIdx.x;

    __shared__ float tr[CC * CC];
    __shared__ float sc[2][CP];

    for (int i = tid; i < CC * CC; i += 64) tr[i] = trans[i];
    if (tid < CC) sc[0][tid] = startv[tid] + ems[((size_t)b * TT) * CP + tid];
    __syncthreads();

    int cur = 0;
    for (int t = 1; t < TT; ++t) {
        if (tid < CC) {
            float best = -3.0e38f;
            int   arg  = 0;
            #pragma unroll 1
            for (int p = 0; p < CC; ++p) {
                float v = sc[cur][p] + tr[p * CC + tid];
                if (v > best) { best = v; arg = p; }
            }
            sc[cur ^ 1][tid] = best + ems[((size_t)b * TT + t) * CP + tid];
            bp[((size_t)b * (TT - 1) + (t - 1)) * CP + tid] = (unsigned char)arg;
        }
        __syncthreads();
        cur ^= 1;
    }

    if (tid == 0) {
        float best = -3.0e38f;
        int   arg  = 0;
        for (int j = 0; j < CC; ++j) {
            float v = sc[cur][j] + endv[j];
            if (v > best) { best = v; arg = j; }
        }
        int tag = arg;
        out[(size_t)b * TT + (TT - 1)] = tag;
        for (int t = TT - 2; t >= 0; --t) {
            tag = bp[((size_t)b * (TT - 1) + t) * CP + tag];
            out[(size_t)b * TT + t] = tag;
        }
    }
}

// ---------------------------------------------------------------------------
// Host launcher
// ---------------------------------------------------------------------------
extern "C" void kernel_launch(void* const* d_in, const int* in_sizes, int n_in,
                              void* d_out, int out_size, void* d_ws, size_t ws_size,
                              hipStream_t stream) {
    (void)in_sizes; (void)n_in; (void)out_size; (void)ws_size;

    const float* x         = (const float*)d_in[0];   // [B,T,39]
    const float* w_ih_l0   = (const float*)d_in[1];   // [2,256,39]
    const float* w_hh_l0   = (const float*)d_in[2];   // [2,256,64]
    const float* b_l0      = (const float*)d_in[3];   // [2,256]
    const float* w_ih_r    = (const float*)d_in[4];   // [2,2,256,128]
    const float* w_hh_r    = (const float*)d_in[5];   // [2,2,256,64]
    const float* b_r       = (const float*)d_in[6];   // [2,2,256]
    const float* lin_w     = (const float*)d_in[7];   // [41,128]
    const float* lin_b     = (const float*)d_in[8];   // [41]
    const float* crf_start = (const float*)d_in[9];   // [41]
    const float* crf_end   = (const float*)d_in[10];  // [41]
    const float* crf_trans = (const float*)d_in[11];  // [41,41]
    int* out = (int*)d_out;

    char* ws = (char*)d_ws;
    size_t off = 0;
    auto alloc = [&](size_t bytes) -> char* {
        char* p = ws + off;
        off = (off + bytes + 255) & ~(size_t)255;
        return p;
    };

    h16*  xpad  = (h16*)alloc((size_t)BB * TT * 64 * 2);        // 32 MB
    h16*  hbuf0 = (h16*)alloc((size_t)BB * TT * 128 * 2);       // 64 MB
    h16*  hbuf1 = (h16*)alloc((size_t)BB * TT * 128 * 2);       // 64 MB
    float* ems  = (float*)alloc((size_t)BB * TT * CP * 4);      // 48 MB
    unsigned char* bp = (unsigned char*)alloc((size_t)BB * (TT - 1) * CP); // 12.6 MB
    h16*  wih0f = (h16*)alloc((size_t)2 * GG * 64 * 2);
    h16*  whh0f = (h16*)alloc((size_t)2 * GG * 64 * 2);
    h16*  wihrf = (h16*)alloc((size_t)2 * 2 * GG * 128 * 2);
    h16*  whhrf = (h16*)alloc((size_t)2 * 2 * GG * 64 * 2);
    h16*  linwf = (h16*)alloc((size_t)CP * 128 * 2);

    // --- weight / input conversion to f16 ---
    {
        int n = BB * TT * 64;
        pad_x_kernel<<<(n + 255) / 256, 256, 0, stream>>>(x, xpad, n);
    }
    {
        int n = 2 * GG * 64;
        pad_wih0_kernel<<<(n + 255) / 256, 256, 0, stream>>>(w_ih_l0, wih0f, n);
        cvt_kernel<<<(n + 255) / 256, 256, 0, stream>>>(w_hh_l0, whh0f, n);
    }
    {
        int n = 2 * 2 * GG * 128;
        cvt_kernel<<<(n + 255) / 256, 256, 0, stream>>>(w_ih_r, wihrf, n);
        n = 2 * 2 * GG * 64;
        cvt_kernel<<<(n + 255) / 256, 256, 0, stream>>>(w_hh_r, whhrf, n);
        pad_linw_kernel<<<(CP * 128 + 255) / 256, 256, 0, stream>>>(lin_w, linwf);
    }

    // --- 3-layer BiLSTM (grid: 16 batch tiles x 2 directions) ---
    dim3 sg(BB / 16, 2);
    lstm_scan_kernel<64><<<sg, 512, 0, stream>>>(xpad, wih0f, whh0f, b_l0, hbuf0);
    lstm_scan_kernel<128><<<sg, 512, 0, stream>>>(hbuf0, wihrf, whhrf, b_r, hbuf1);
    lstm_scan_kernel<128><<<sg, 512, 0, stream>>>(
        hbuf1, wihrf + (size_t)2 * GG * 128, whhrf + (size_t)2 * GG * 64,
        b_r + 2 * GG, hbuf0);

    // --- emissions GEMM (49152 wave-tasks / 8 per block) ---
    emis_kernel<<<(BB * TT / 16) * 3 / 8, 256, 0, stream>>>(hbuf0, linwf, lin_b, ems);

    // --- CRF Viterbi decode ---
    viterbi_kernel<<<BB, 64, 0, stream>>>(ems, crf_start, crf_end, crf_trans, bp, out);
}
